// UnidirectionalLSTMSentenceEncoder_8813272891997
// MI455X (gfx1250) — matrix operand order repaired and checked by reading, not compiled, not run on hardware
//
#include <hip/hip_runtime.h>
#include <math.h>

// ---------------------------------------------------------------------------
// UnidirectionalLSTMSentenceEncoder for MI455X (gfx1250, wave32, WMMA).
//
// Sizes: B=64, T=512, E=1024, H=2048, 4H=8192.
//
// The 512-step recurrence is L2-bandwidth bound (bf16 weights, 48MB/step,
// resident in the 192MB L2). Per step: one kernel, 128 WGs x 256 threads.
// WG g owns hidden cols [16g,16g+16); its 8 waves = 4 gates x 2 M-halves,
// each wave computing a 32x16 C-strip via v_wmma_f32_16x16x32_bf16 over a
// unified K = E + H = 3072 loop (x-projection fused into the step; avoids a
// 1GiB xg buffer). Software pipelining: next k-block's weight B-fragments
// and activation panel are fetched to registers while the current block's
// WMMAs run; the LDS A panel is double-buffered with one barrier per block.
// A-fragments are batch-loaded (one s_wait_dscnt per block), and the block
// loop is unrolled x2 so LDS buffer parity is static and B-fragment
// rotation becomes register ping-pong instead of v_mov chains.
// ---------------------------------------------------------------------------

#define B_   64
#define T_   512
#define E_   1024
#define H_   2048
#define G4H  (4 * H_)
#define KTOT (E_ + H_)          // 3072 combined K
#define NB   (KTOT / 64)        // 48 k-blocks of 64

typedef __attribute__((ext_vector_type(16))) __bf16        v16bf;
typedef __attribute__((ext_vector_type(8)))  float         v8f;
typedef __attribute__((ext_vector_type(4)))  unsigned int  u32x4;
typedef __attribute__((ext_vector_type(4)))  float         f32x4;

union Frag16 { v16bf v; u32x4 q[2]; };

__device__ __forceinline__ unsigned short f2bf(float f) {
  unsigned int u = __float_as_uint(f);
  u += 0x7fffu + ((u >> 16) & 1u);
  return (unsigned short)(u >> 16);
}

// ---------------- workspace layout (bytes), total ~48.8 MB ------------------
static const size_t OFF_WIH  = 0;                                   // 16.0 MB bf16 W_ih
static const size_t OFF_WHH  = OFF_WIH + (size_t)G4H * E_ * 2;      // 32.0 MB bf16 W_hh
static const size_t OFF_BIAS = OFF_WHH + (size_t)G4H * H_ * 2;      // 32 KB fused bias
static const size_t OFF_H    = OFF_BIAS + (size_t)G4H * 4;          // 256 KB bf16 h
static const size_t OFF_C    = OFF_H + (size_t)B_ * H_ * 2;         // 512 KB f32 c

// ---------------------------------------------------------------------------
__global__ __launch_bounds__(256)
void lstm_init_kernel(const float* __restrict__ wih_f,
                      const float* __restrict__ whh_f,
                      const float* __restrict__ b_ih,
                      const float* __restrict__ b_hh,
                      unsigned short* __restrict__ wih,
                      unsigned short* __restrict__ whh,
                      float* __restrict__ bias,
                      unsigned short* __restrict__ hbf,
                      float* __restrict__ cst) {
  size_t i = (size_t)blockIdx.x * blockDim.x + threadIdx.x;
  const size_t stride = (size_t)gridDim.x * blockDim.x;
  const size_t NHH = (size_t)G4H * H_;
  const size_t NIH = (size_t)G4H * E_;
  for (; i < NHH; i += stride) {
    whh[i] = f2bf(whh_f[i]);
    if (i < NIH)              wih[i] = f2bf(wih_f[i]);
    if (i < (size_t)G4H)      bias[i] = b_ih[i] + b_hh[i];
    if (i < (size_t)B_ * H_)  { hbf[i] = 0; cst[i] = 0.0f; }
  }
}

// ---------------------------------------------------------------------------
__global__ __launch_bounds__(256)
void lstm_step_kernel(const float* __restrict__ x,
                      const int*   __restrict__ lens,
                      const unsigned short* __restrict__ wih,
                      const unsigned short* __restrict__ whh,
                      const float* __restrict__ bias,
                      unsigned short* __restrict__ hbf,
                      float* __restrict__ cst,
                      float* __restrict__ out,
                      int t) {
  // Double-buffered activation panel: 2 x (64 rows x 64 cols bf16, row pad 8
  // -> 144B row stride kills the 16-row bank conflict on fragment reads).
  __shared__ unsigned short sA[2 * 64 * 72];   // 18 KB
  __shared__ float          sG[4 * 64 * 16];   // 16 KB gate tiles (i,f,g,o)

  const int tid   = threadIdx.x;
  const int lane  = tid & 31;
  const int lm    = lane & 15;        // N-col within tile / M-row within tile
  const int lh    = lane >> 4;        // K-half selector
  const int wid   = tid >> 5;         // wave 0..7
  const int gate  = wid >> 1;         // 0:i 1:f 2:g 3:o
  const int mhalf = wid & 1;          // which 32-row half of M=64
  const int colbase = blockIdx.x * 16;

  // staging role: 4 threads per activation row, 16 cols each
  const int srow = tid >> 2;          // 0..63 batch row
  const int sq   = tid & 3;           // 0..3 16-col quarter

  v8f acc[2] = {v8f{0.f,0.f,0.f,0.f,0.f,0.f,0.f,0.f},
                v8f{0.f,0.f,0.f,0.f,0.f,0.f,0.f,0.f}};

  const int nrow = gate * H_ + colbase + lm;   // weight row = gate output col
  const unsigned short* pw_ih = wih + (size_t)nrow * E_ + lh * 16;
  const unsigned short* pw_hh = whh + (size_t)nrow * H_ + lh * 16;
  const float*          xrow  = x + ((size_t)srow * T_ + t) * E_ + sq * 16;
  const unsigned short* hrow  = hbf + (size_t)srow * H_ + sq * 16;

  // ---- prologue: B-frags + staged panel for k-block 0 (x phase) ----------
  Frag16 bcur[2];
  {
    const u32x4* p0 = (const u32x4*)pw_ih;
    const u32x4* p1 = (const u32x4*)(pw_ih + 32);
    bcur[0].q[0] = p0[0]; bcur[0].q[1] = p0[1];
    bcur[1].q[0] = p1[0]; bcur[1].q[1] = p1[1];

    const f32x4* px = (const f32x4*)xrow;
    unsigned short tmp[16];
#pragma unroll
    for (int v = 0; v < 4; ++v) {
      f32x4 d = px[v];
      tmp[v*4+0] = f2bf(d[0]); tmp[v*4+1] = f2bf(d[1]);
      tmp[v*4+2] = f2bf(d[2]); tmp[v*4+3] = f2bf(d[3]);
    }
    u32x4* dst = (u32x4*)&sA[srow * 72 + sq * 16];
    dst[0] = *(const u32x4*)&tmp[0];
    dst[1] = *(const u32x4*)&tmp[8];
  }
  __syncthreads();

  // ---- main pipelined loop over 48 k-blocks ------------------------------
#pragma unroll 2
  for (int i = 0; i < NB; ++i) {
    const int nkb = (i + 1) * 64;     // next block's combined-K base (uniform)
    Frag16 bnext[2];
    f32x4 sx0, sx1, sx2, sx3;         // staged next panel (x phase)
    u32x4 sh0, sh1;                   // staged next panel (h phase)
    const bool have_next = (i + 1 < NB);
    const bool next_is_x = (nkb < E_);

    if (have_next) {
      // issue next block's weight + activation loads first (latency hiding)
      const unsigned short* pn = next_is_x ? (pw_ih + nkb)
                                           : (pw_hh + (nkb - E_));
      const u32x4* p0 = (const u32x4*)pn;
      const u32x4* p1 = (const u32x4*)(pn + 32);
      bnext[0].q[0] = p0[0]; bnext[0].q[1] = p0[1];
      bnext[1].q[0] = p1[0]; bnext[1].q[1] = p1[1];
      if (next_is_x) {
        const f32x4* px = (const f32x4*)(xrow + nkb);
        sx0 = px[0]; sx1 = px[1]; sx2 = px[2]; sx3 = px[3];
      } else {
        const u32x4* ph = (const u32x4*)(hrow + (nkb - E_));
        sh0 = ph[0]; sh1 = ph[1];
      }
    }

    // compute current block: batch-load all 4 A-fragments, then 4 WMMAs
    const unsigned short* bufA = &sA[(i & 1) * 64 * 72];
    Frag16 af[2][2];
#pragma unroll
    for (int s = 0; s < 2; ++s) {
#pragma unroll
      for (int mt = 0; mt < 2; ++mt) {
        const unsigned short* pa =
            &bufA[((mhalf * 2 + mt) * 16 + lm) * 72 + s * 32 + lh * 16];
        af[s][mt].q[0] = *(const u32x4*)pa;
        af[s][mt].q[1] = *(const u32x4*)(pa + 8);
      }
    }
#pragma unroll
    for (int s = 0; s < 2; ++s) {
#pragma unroll
      for (int mt = 0; mt < 2; ++mt) {
        acc[mt] = __builtin_amdgcn_wmma_f32_16x16x32_bf16(
            false, af[s][mt].v, false, bcur[s].v, (short)0, acc[mt],
            false, false);
      }
    }

    // commit next panel to the other LDS buffer, rotate B frags
    if (have_next) {
      u32x4 d0, d1;
      if (next_is_x) {
        unsigned short tmp[16];
        f32x4 sv[4] = {sx0, sx1, sx2, sx3};
#pragma unroll
        for (int v = 0; v < 4; ++v) {
          tmp[v*4+0] = f2bf(sv[v][0]); tmp[v*4+1] = f2bf(sv[v][1]);
          tmp[v*4+2] = f2bf(sv[v][2]); tmp[v*4+3] = f2bf(sv[v][3]);
        }
        d0 = *(const u32x4*)&tmp[0];
        d1 = *(const u32x4*)&tmp[8];
      } else {
        d0 = sh0; d1 = sh1;
      }
      u32x4* dst = (u32x4*)&sA[((i + 1) & 1) * 64 * 72 + srow * 72 + sq * 16];
      dst[0] = d0;
      dst[1] = d1;
      bcur[0] = bnext[0];
      bcur[1] = bnext[1];
    }
    __syncthreads();
  }

  // ---- dump gate tiles to LDS --------------------------------------------
  // C layout: VGPR r, lane L -> M = r + 8*(L/16), N = L%16
#pragma unroll
  for (int mt = 0; mt < 2; ++mt) {
#pragma unroll
    for (int r = 0; r < 8; ++r) {
      const int M = mhalf * 32 + mt * 16 + r + 8 * lh;
      sG[(gate * 64 + M) * 16 + lm] = acc[mt][r];
    }
  }
  __syncthreads();

  // ---- LSTM pointwise: 64 rows x 16 cols = 1024 cells, 4 per thread ------
  {
    const int idx0 = tid * 4;
    const int m    = idx0 >> 4;        // batch row
    const int c0   = idx0 & 15;        // col within WG slice
    const int last = lens[m] - 1;
#pragma unroll
    for (int j = 0; j < 4; ++j) {
      const int c   = c0 + j;
      const int col = colbase + c;
      float ig = sG[(0 * 64 + m) * 16 + c] + bias[0 * H_ + col];
      float fg = sG[(1 * 64 + m) * 16 + c] + bias[1 * H_ + col];
      float gg = sG[(2 * 64 + m) * 16 + c] + bias[2 * H_ + col];
      float og = sG[(3 * 64 + m) * 16 + c] + bias[3 * H_ + col];
      float is = 1.0f / (1.0f + __expf(-ig));
      float fs = 1.0f / (1.0f + __expf(-fg));
      float gt = tanhf(gg);
      float os = 1.0f / (1.0f + __expf(-og));
      const size_t gidx = (size_t)m * H_ + col;
      float cn = fs * cst[gidx] + is * gt;
      cst[gidx] = cn;
      float hn = os * tanhf(cn);
      hbf[gidx] = f2bf(hn);
      if (last == t) out[gidx] = hn;   // exactly one write per (b,col) per call
    }
  }
}

// ---------------------------------------------------------------------------
extern "C" void kernel_launch(void* const* d_in, const int* in_sizes, int n_in,
                              void* d_out, int out_size, void* d_ws, size_t ws_size,
                              hipStream_t stream) {
  const float* x    = (const float*)d_in[0];   // [B, T, E]
  const int*   lens = (const int*)d_in[1];     // [B]
  const float* Wih  = (const float*)d_in[2];   // [4H, E]
  const float* Whh  = (const float*)d_in[3];   // [4H, H]
  const float* bih  = (const float*)d_in[4];   // [4H]
  const float* bhh  = (const float*)d_in[5];   // [4H]
  float* out = (float*)d_out;                  // [B, H]

  char* ws = (char*)d_ws;                      // needs ~48.8 MB
  unsigned short* wih_bf = (unsigned short*)(ws + OFF_WIH);
  unsigned short* whh_bf = (unsigned short*)(ws + OFF_WHH);
  float*          bias   = (float*)(ws + OFF_BIAS);
  unsigned short* hbf    = (unsigned short*)(ws + OFF_H);
  float*          cst    = (float*)(ws + OFF_C);

  lstm_init_kernel<<<8192, 256, 0, stream>>>(Wih, Whh, bih, bhh,
                                             wih_bf, whh_bf, bias, hbf, cst);

  for (int t = 0; t < T_; ++t) {
    lstm_step_kernel<<<128, 256, 0, stream>>>(x, lens, wih_bf, whh_bf, bias,
                                              hbf, cst, out, t);
  }
}
